// Hgru2_2d_parallel_67559835566284
// MI455X (gfx1250) — matrix operand order repaired and checked by reading, not compiled, not run on hardware
//
#include <hip/hip_runtime.h>

typedef __attribute__((ext_vector_type(16))) _Float16 v16h;
typedef __attribute__((ext_vector_type(8)))  float    v8f;

// Problem constants (fixed by reference setup)
constexpr int HH  = 64;
constexpr int WW  = 64;
constexpr int BB  = 8;
constexpr int DD  = 256;
constexpr int MM  = HH * WW * BB;   // 32768 rows
constexpr int KK  = DD;             // 256 (GEMM K for both)
constexpr int N1  = 3 * DD;         // 768
constexpr int N2  = DD;             // 256
constexpr int CC  = 2 * DD;         // 512 scan channels
constexpr int NH  = DD / 2;         // 128 heads, E=2

// ---------------------------------------------------------------------------
// async global -> LDS copy (gfx1250), tracked on ASYNCcnt.
// INST_OFFSET applies to BOTH the LDS destination and the global source, so a
// single base pair + immediate offset covers a whole contiguous per-thread run.
// ---------------------------------------------------------------------------
#define ASYNC_B128(ldsoff, gptr, OFF)                                          \
    asm volatile("global_load_async_to_lds_b128 %0, %1, off offset:" #OFF     \
                 :: "v"(ldsoff), "v"(gptr) : "memory")

// ---------------------------------------------------------------------------
// f32 -> f16 conversion (row-major copy)
// ---------------------------------------------------------------------------
__global__ __launch_bounds__(256) void f32_to_f16_kernel(const float* __restrict__ in,
                                                         _Float16* __restrict__ out,
                                                         int n) {
    int i = blockIdx.x * 256 + threadIdx.x;
    if (i < n) out[i] = (_Float16)in[i];
}

// f32 (R x C row-major) -> f16 transposed (C x R row-major)
__global__ __launch_bounds__(256) void f32_to_f16_t_kernel(const float* __restrict__ in,
                                                           _Float16* __restrict__ out,
                                                           int R, int Ncols) {
    int i = blockIdx.x * 256 + threadIdx.x;
    if (i < R * Ncols) {
        int r = i / Ncols, c = i % Ncols;
        out[(size_t)c * R + r] = (_Float16)in[i];
    }
}

// ---------------------------------------------------------------------------
// 128-wide-K tile compute: 16 WMMAs, distinct fragment registers so the
// scheduler can overlap ds_load latency with matrix ops.
// ---------------------------------------------------------------------------
__device__ __forceinline__ void compute_tile(const _Float16 (&tA)[128][128],
                                             const _Float16 (&tB)[64][128],
                                             int am, int half, int l16, v8f acc[4]) {
#pragma unroll
    for (int k0 = 0; k0 < 128; k0 += 32) {
        union F { v16h v; uint4 q[2]; };
        F af, b0, b1, b2, b3;
        // A fragment (contiguous K order, same as B):
        // elems i <-> K = 16*half + i
        af.q[0] = *(const uint4*)(&tA[am][k0 + half * 16]);
        af.q[1] = *(const uint4*)(&tA[am][k0 + half * 16 + 8]);
        // B fragments: lane -> N = l16 (+16j), elems i <-> K = 16*half + i
        b0.q[0] = *(const uint4*)(&tB[l16][k0 + half * 16]);
        b0.q[1] = *(const uint4*)(&tB[l16][k0 + half * 16 + 8]);
        b1.q[0] = *(const uint4*)(&tB[16 + l16][k0 + half * 16]);
        b1.q[1] = *(const uint4*)(&tB[16 + l16][k0 + half * 16 + 8]);
        b2.q[0] = *(const uint4*)(&tB[32 + l16][k0 + half * 16]);
        b2.q[1] = *(const uint4*)(&tB[32 + l16][k0 + half * 16 + 8]);
        b3.q[0] = *(const uint4*)(&tB[48 + l16][k0 + half * 16]);
        b3.q[1] = *(const uint4*)(&tB[48 + l16][k0 + half * 16 + 8]);
        acc[0] = __builtin_amdgcn_wmma_f32_16x16x32_f16(false, af.v, false, b0.v, (short)0, acc[0], false, false);
        acc[1] = __builtin_amdgcn_wmma_f32_16x16x32_f16(false, af.v, false, b1.v, (short)0, acc[1], false, false);
        acc[2] = __builtin_amdgcn_wmma_f32_16x16x32_f16(false, af.v, false, b2.v, (short)0, acc[2], false, false);
        acc[3] = __builtin_amdgcn_wmma_f32_16x16x32_f16(false, af.v, false, b3.v, (short)0, acc[3], false, false);
    }
}

// ---------------------------------------------------------------------------
// WMMA f16 GEMM:  C[M x N] = A[M x K] * Bt[N x K]^T + bias[N]
// Block tile 128x64, K=256 as two 128-wide double-buffered async-DMA stages.
// 256 threads = 8 waves, wave tile 16x64 (4 accumulators).
// ---------------------------------------------------------------------------
__global__ __launch_bounds__(256) void gemm_f16_wmma_kernel(
        const _Float16* __restrict__ A, const _Float16* __restrict__ Bt,
        const float* __restrict__ bias, float* __restrict__ C, int N) {
    constexpr int BM = 128, BN = 64, BK = 128;
    __shared__ _Float16 sA[2][BM][BK];   // 64 KB
    __shared__ _Float16 sB[2][BN][BK];   // 32 KB

    const int tid  = threadIdx.x;
    const int wave = tid >> 5;
    const int lane = tid & 31;
    const int half = lane >> 4;
    const int l16  = lane & 15;

    const int rowBase = blockIdx.x * BM;
    const int colBase = blockIdx.y * BN;

    // per-thread staging coordinates (contiguous 128B of A, 64B of B per stage)
    const int ar = tid >> 1, ac = (tid & 1) * 64;    // halves
    const int bn = tid >> 2, bc = (tid & 3) * 32;    // halves
    const _Float16* gA = A  + (size_t)(rowBase + ar) * KK + ac;
    const _Float16* gB = Bt + (size_t)(colBase + bn) * KK + bc;
    const unsigned lA0 = (unsigned)(uintptr_t)(&sA[0][ar][ac]);
    const unsigned lB0 = (unsigned)(uintptr_t)(&sB[0][bn][bc]);
    const unsigned lA1 = (unsigned)(uintptr_t)(&sA[1][ar][ac]);
    const unsigned lB1 = (unsigned)(uintptr_t)(&sB[1][bn][bc]);

    // ---- issue BOTH K-stages' DMA up front (12 async copies per stage) ----
    ASYNC_B128(lA0, gA, 0);   ASYNC_B128(lA0, gA, 16);
    ASYNC_B128(lA0, gA, 32);  ASYNC_B128(lA0, gA, 48);
    ASYNC_B128(lA0, gA, 64);  ASYNC_B128(lA0, gA, 80);
    ASYNC_B128(lA0, gA, 96);  ASYNC_B128(lA0, gA, 112);
    ASYNC_B128(lB0, gB, 0);   ASYNC_B128(lB0, gB, 16);
    ASYNC_B128(lB0, gB, 32);  ASYNC_B128(lB0, gB, 48);
    const _Float16* gA1 = gA + BK;
    const _Float16* gB1 = gB + BK;
    ASYNC_B128(lA1, gA1, 0);  ASYNC_B128(lA1, gA1, 16);
    ASYNC_B128(lA1, gA1, 32); ASYNC_B128(lA1, gA1, 48);
    ASYNC_B128(lA1, gA1, 64); ASYNC_B128(lA1, gA1, 80);
    ASYNC_B128(lA1, gA1, 96); ASYNC_B128(lA1, gA1, 112);
    ASYNC_B128(lB1, gB1, 0);  ASYNC_B128(lB1, gB1, 16);
    ASYNC_B128(lB1, gB1, 32); ASYNC_B128(lB1, gB1, 48);

    // accumulators seeded with bias (bias depends only on column = l16)
    v8f acc[4];
#pragma unroll
    for (int j = 0; j < 4; ++j) {
        float b = bias[colBase + j * 16 + l16];
#pragma unroll
        for (int r = 0; r < 8; ++r) acc[j][r] = b;
    }

    const int am = wave * 16 + l16;

    // ---- stage 0 ready (async loads complete in order: oldest 12 done) ----
    asm volatile("s_wait_asynccnt 12" ::: "memory");
    __syncthreads();
    compute_tile(sA[0], sB[0], am, half, l16, acc);   // stage-1 DMA still in flight

    // ---- stage 1 ready ----
    asm volatile("s_wait_asynccnt 0" ::: "memory");
    __syncthreads();
    compute_tile(sA[1], sB[1], am, half, l16, acc);

    // ---- store: C layout -> row = 8*half + r, col = l16 ----
#pragma unroll
    for (int j = 0; j < 4; ++j) {
        int col = colBase + j * 16 + l16;
#pragma unroll
        for (int r = 0; r < 8; ++r) {
            int row = rowBase + wave * 16 + half * 8 + r;
            C[(size_t)row * N + col] = acc[j][r];
        }
    }
}

// ---------------------------------------------------------------------------
// Pointwise: silu/sigmoid gates + (1-lambda) (x) input outer product
// ---------------------------------------------------------------------------
__global__ __launch_bounds__(256) void pointwise_kernel(
        const float* __restrict__ feat, float* __restrict__ inp_f,
        float* __restrict__ lam_f, float* __restrict__ og_buf) {
    size_t t = (size_t)blockIdx.x * 256 + threadIdx.x;   // r*NH + h
    if (t >= (size_t)MM * NH) return;
    int h = (int)(t & (NH - 1));
    size_t r = t >> 7;                                   // NH = 128
    const float* row = feat + r * N1;
    float i0 = row[h * 2],       i1 = row[h * 2 + 1];
    float g0 = row[DD + h * 2],  g1 = row[DD + h * 2 + 1];
    float f0 = row[2*DD + h*2],  f1 = row[2*DD + h*2 + 1];
    float si0 = i0 / (1.f + __expf(-i0));
    float si1 = i1 / (1.f + __expf(-i1));
    float l0  = 1.f / (1.f + __expf(-f0));
    float l1  = 1.f / (1.f + __expf(-f1));
    float o0  = 1.f / (1.f + __expf(-g0));
    float o1  = 1.f / (1.f + __expf(-g1));
    float* ip = inp_f + r * CC + h * 4;
    float* lp = lam_f + r * CC + h * 4;
    ip[0] = (1.f - l0) * si0; ip[1] = (1.f - l0) * si1;   // d=0, e=0/1
    ip[2] = (1.f - l1) * si0; ip[3] = (1.f - l1) * si1;   // d=1, e=0/1
    lp[0] = l0; lp[1] = l0; lp[2] = l1; lp[3] = l1;
    og_buf[r * DD + h * 2]     = o0;
    og_buf[r * DD + h * 2 + 1] = o1;
}

// ---------------------------------------------------------------------------
// Bidirectional scan along H: forward partials in VGPRs, single store.
// ---------------------------------------------------------------------------
__global__ __launch_bounds__(256) void scan_h_kernel(
        const float* __restrict__ inp, const float* __restrict__ lam,
        float* __restrict__ state) {
    size_t t = (size_t)blockIdx.x * 256 + threadIdx.x;
    const size_t NSEQ = (size_t)WW * BB * CC;            // 262144
    if (t >= NSEQ) return;
    const size_t stride = NSEQ;
    float fwd[HH];
    float acc = 0.f;
    size_t idx = t;
#pragma unroll
    for (int h = 0; h < HH; ++h) {
        acc = fmaf(lam[idx], acc, inp[idx]);
        fwd[h] = acc;
        idx += stride;
    }
    acc = 0.f;
    idx = t + (size_t)(HH - 1) * stride;
#pragma unroll
    for (int h = HH - 1; h >= 0; --h) {
        acc = fmaf(lam[idx], acc, inp[idx]);
        state[idx] = fwd[h] + acc;
        idx -= stride;
    }
}

// ---------------------------------------------------------------------------
// Bidirectional scan along W: accumulates into state written by scan_h.
// ---------------------------------------------------------------------------
__global__ __launch_bounds__(256) void scan_w_kernel(
        const float* __restrict__ inp, const float* __restrict__ lam,
        float* __restrict__ state) {
    size_t t = (size_t)blockIdx.x * 256 + threadIdx.x;
    const size_t NSEQ = (size_t)HH * BB * CC;            // 262144
    if (t >= NSEQ) return;
    size_t h   = t >> 12;                                // BB*CC = 4096
    size_t rem = t & 4095;
    size_t base = h * ((size_t)WW * BB * CC) + rem;
    const size_t stride = (size_t)BB * CC;               // 4096
    float fwd[WW];
    float acc = 0.f;
    size_t idx = base;
#pragma unroll
    for (int w = 0; w < WW; ++w) {
        acc = fmaf(lam[idx], acc, inp[idx]);
        fwd[w] = acc;
        idx += stride;
    }
    acc = 0.f;
    idx = base + (size_t)(WW - 1) * stride;
#pragma unroll
    for (int w = WW - 1; w >= 0; --w) {
        acc = fmaf(lam[idx], acc, inp[idx]);
        state[idx] += fwd[w] + acc;                      // add onto H-scan result
        idx -= stride;
    }
}

// ---------------------------------------------------------------------------
// Output-gate contraction + LayerNorm over D, emit f16 for GEMM2
// ---------------------------------------------------------------------------
__global__ __launch_bounds__(256) void gate_ln_kernel(
        const float* __restrict__ state, const float* __restrict__ og,
        const float* __restrict__ gamma, const float* __restrict__ beta,
        _Float16* __restrict__ out16) {
    __shared__ float red[18];
    size_t r = blockIdx.x;
    int d = threadIdx.x;           // output channel = h*2 + e
    int h = d >> 1, e = d & 1;
    const float* srow = state + r * CC;
    const float* orow = og + r * DD;
    float val = srow[h * 4 + e] * orow[h * 2] + srow[h * 4 + 2 + e] * orow[h * 2 + 1];

    float s = val, q = val * val;
#pragma unroll
    for (int off = 16; off > 0; off >>= 1) {
        s += __shfl_xor(s, off, 32);
        q += __shfl_xor(q, off, 32);
    }
    int wv = threadIdx.x >> 5, lane = threadIdx.x & 31;
    if (lane == 0) { red[wv] = s; red[8 + wv] = q; }
    __syncthreads();
    if (threadIdx.x == 0) {
        float ts = 0.f, tq = 0.f;
#pragma unroll
        for (int i = 0; i < 8; ++i) { ts += red[i]; tq += red[8 + i]; }
        float mu = ts * (1.f / DD);
        red[16] = mu;
        red[17] = rsqrtf(tq * (1.f / DD) - mu * mu + 1e-5f);
    }
    __syncthreads();
    float mu = red[16], rstd = red[17];
    float y = (val - mu) * rstd * gamma[d] + beta[d];
    out16[r * DD + d] = (_Float16)y;
}

// ---------------------------------------------------------------------------
extern "C" void kernel_launch(void* const* d_in, const int* in_sizes, int n_in,
                              void* d_out, int out_size, void* d_ws, size_t ws_size,
                              hipStream_t stream) {
    const float* x     = (const float*)d_in[0];   // (H,W,B,D)
    const float* in_w  = (const float*)d_in[1];   // (D, 3D)
    const float* in_b  = (const float*)d_in[2];   // (3D)
    const float* out_w = (const float*)d_in[3];   // (D, D)
    const float* out_b = (const float*)d_in[4];   // (D)
    const float* gamma = (const float*)d_in[5];
    const float* beta  = (const float*)d_in[6];
    float* out = (float*)d_out;

    char* ws = (char*)d_ws;
    size_t off = 0;
    auto carve = [&](size_t bytes) {
        size_t o = off;
        off += (bytes + 255) & ~(size_t)255;
        return o;
    };
    // f16 copies (weights stored TRANSPOSED: rows = N, contiguous K)
    _Float16* xh   = (_Float16*)(ws + carve((size_t)MM * KK * 2));       // 16 MB
    _Float16* iwT  = (_Float16*)(ws + carve((size_t)KK * N1 * 2));       // 384 KB
    _Float16* owT  = (_Float16*)(ws + carve((size_t)KK * N2 * 2));       // 128 KB
    // feat (96 MB) — state (64 MB) overlays it after pointwise consumes feat
    char* featRegion = ws + carve((size_t)MM * N1 * 4);
    float* feat  = (float*)featRegion;
    float* state = (float*)featRegion;
    // inp_f (64 MB) — ln f16 output (16 MB) overlays it after scans are done
    char* inpRegion = ws + carve((size_t)MM * CC * 4);
    float*    inp_f = (float*)inpRegion;
    _Float16* lnh   = (_Float16*)inpRegion;
    float* lam_f = (float*)(ws + carve((size_t)MM * CC * 4));            // 64 MB
    float* ogb   = (float*)(ws + carve((size_t)MM * DD * 4));            // 32 MB

    // 1) precision conversion for tensor-core inputs (weights also transposed)
    f32_to_f16_kernel<<<(MM * KK + 255) / 256, 256, 0, stream>>>(x, xh, MM * KK);
    f32_to_f16_t_kernel<<<(KK * N1 + 255) / 256, 256, 0, stream>>>(in_w, iwT, KK, N1);
    f32_to_f16_t_kernel<<<(KK * N2 + 255) / 256, 256, 0, stream>>>(out_w, owT, KK, N2);

    // 2) feat = x @ in_w + in_b   (WMMA, double-buffered async-to-LDS DMA)
    gemm_f16_wmma_kernel<<<dim3(MM / 128, N1 / 64), 256, 0, stream>>>(xh, iwT, in_b, feat, N1);

    // 3) gates + outer products
    pointwise_kernel<<<(MM * NH) / 256, 256, 0, stream>>>(feat, inp_f, lam_f, ogb);

    // 4) bidirectional scans (H then W, accumulated into state)
    scan_h_kernel<<<(WW * BB * CC) / 256, 256, 0, stream>>>(inp_f, lam_f, state);
    scan_w_kernel<<<(HH * BB * CC) / 256, 256, 0, stream>>>(inp_f, lam_f, state);

    // 5) output gate contraction + LayerNorm -> f16
    gate_ln_kernel<<<MM, 256, 0, stream>>>(state, ogb, gamma, beta, lnh);

    // 6) final projection: out = ln @ out_w + out_b   (WMMA)
    gemm_f16_wmma_kernel<<<dim3(MM / 128, N2 / 64), 256, 0, stream>>>(lnh, owT, out_b, out, N2);
}
